// EdgeLayer_51513837748550
// MI455X (gfx1250) — matrix operand order repaired and checked by reading, not compiled, not run on hardware
//
#include <hip/hip_runtime.h>

// ---------------------------------------------------------------------------
// M3GNet EdgeLayer for MI455X (gfx1250, wave32, WMMA f32_16x16x32_f16).
// feat[E,320] = [atom[src] | atom[dst] | prime]; GatedMLP 320->128->64->128
// (silu main path, silu/sigmoid gate path), per-edge scalar gate, residual.
// All weights transposed->f16 in LDS (229 KB); per-wave 16-edge tiles.
// Round 2: zero-init accumulators (bias folded into epilogue), fused dual
// gemm for layer 0 (A-frag loaded once), fully coalesced float4 epilogue
// through an LDS transpose pass, shfl-broadcast edge indices.
// ---------------------------------------------------------------------------

typedef __attribute__((ext_vector_type(16))) _Float16 v16h;
typedef __attribute__((ext_vector_type(8)))  float    v8f;

union Frag16 { v16h v; uint4 q[2]; };

#define N_EDGES   500000
#define N_NODES   50000
#define TILES     (N_EDGES / 16)        // 31250, exact
#define WAVES_PER_BLOCK 8

// ---- LDS layout (bytes) ----------------------------------------------------
#define OFF_W0T   0                      // f16 [128][320]  (N-major, transposed)
#define OFF_G0T   81920                  // f16 [128][320]
#define OFF_W1T   163840                 // f16 [64][128]
#define OFF_G1T   180224                 // f16 [64][128]
#define OFF_W2T   196608                 // f16 [128][64]
#define OFF_G2T   212992                 // f16 [128][64]
#define OFF_B0    229376                 // f32 [128]
#define OFF_GB0   229888                 // f32 [128]
#define OFF_B1    230400                 // f32 [64]
#define OFF_GB1   230656                 // f32 [64]
#define OFF_B2    230912                 // f32 [128]
#define OFF_GB2   231424                 // f32 [128]
#define OFF_EW    231936                 // f32 [128]
#define OFF_EB    232448                 // f32 [1] (padded to 64B)
#define OFF_STG   232512                 // per-wave staging slabs
#define STG_SIZE  10304                  // 16x320 f16 feat (10240) + s[16] f32 (64)
#define SMEM_TOTAL (OFF_STG + WAVES_PER_BLOCK * STG_SIZE)   // 314944 < 320KB

__device__ __forceinline__ float sigmoidf_(float x) { return 1.0f / (1.0f + __expf(-x)); }
__device__ __forceinline__ float siluf_(float x)    { return x * sigmoidf_(x); }

__device__ __forceinline__ void store2h(_Float16* p, float2 v) {
    union { _Float16 h[2]; unsigned u; } t;
    t.h[0] = (_Float16)v.x; t.h[1] = (_Float16)v.y;
    *(unsigned*)p = t.u;
}

__device__ __forceinline__ void store4h(_Float16* p, float4 v) {
    union { _Float16 h[4]; unsigned long long u; } t;
    t.h[0] = (_Float16)v.x; t.h[1] = (_Float16)v.y;
    t.h[2] = (_Float16)v.z; t.h[3] = (_Float16)v.w;
    *(unsigned long long*)p = t.u;
}

// Transpose f32 [K][N] global weight into f16 [N][K] LDS (one-time per block).
template <int K, int N>
__device__ __forceinline__ void load_wt(_Float16* dst, const float* __restrict__ src,
                                        int tid, int nth) {
    for (int i = tid; i < K * N; i += nth) {
        int n = i / K;            // K is a compile-time constant -> cheap
        int k = i - n * K;
        dst[i] = (_Float16)src[k * N + n];
    }
}

__device__ __forceinline__ void load_vec(float* dst, const float* __restrict__ src,
                                         int n, int tid, int nth) {
    for (int i = tid; i < n; i += nth) dst[i] = src[i];
}

// ---- A fragment (ISA 7.12.2, 16-bit 16x32): lane holds row lane&15; the two
// 8-half groups (K 0..7 / 8..15 selected by lane>=16, then K+16) are each one
// contiguous ds_load_b128 because A is row-major in LDS. -----------------------
__device__ __forceinline__ Frag16 load_a(const _Float16* pa, int kh) {
    Frag16 a;
    a.q[0] = *(const uint4*)(pa + kh);
    a.q[1] = *(const uint4*)(pa + 16 + kh);
    return a;
}
// ---- B fragment (32x16): lane holds col lane&15, two 8-half K groups
// (K 0..7 / 8..15 selected by lane>=16, then K+16), same order as A. ----------
__device__ __forceinline__ Frag16 load_b(const _Float16* pb) {
    Frag16 b;
    b.q[0] = *(const uint4*)(pb);
    b.q[1] = *(const uint4*)(pb + 16);
    return b;
}

// Single-input GEMM: acc[NT] += A[16xK] * W[KxNT*16]; acc zero-initialized so
// the first WMMA of each chain can take SRC2 = inline 0.
template <int K, int NT>
__device__ __forceinline__ void gemm_tile(const _Float16* __restrict__ A, int lda,
                                          const _Float16* __restrict__ Wt,
                                          int lane, v8f (&acc)[NT]) {
    const int col = lane & 15;
    const int kh  = (lane >> 4) * 8;
    const int kb  = (lane >> 4) * 8;
#pragma unroll
    for (int nt = 0; nt < NT; ++nt)
#pragma unroll
        for (int j = 0; j < 8; ++j) acc[nt][j] = 0.0f;
#pragma unroll
    for (int kc = 0; kc < K / 32; ++kc) {
        Frag16 a = load_a(A + (lane & 15) * lda + kc * 32, kh);
#pragma unroll
        for (int nt = 0; nt < NT; ++nt) {
            Frag16 b = load_b(Wt + (nt * 16 + col) * K + kc * 32 + kb);
            acc[nt] = __builtin_amdgcn_wmma_f32_16x16x32_f16(
                false, a.v, false, b.v, (short)0, acc[nt], false, false);
        }
    }
}

// Dual GEMM sharing the A operand (layer 0: main + gate read the same feat).
template <int K, int NT>
__device__ __forceinline__ void gemm_tile_dual(const _Float16* __restrict__ A, int lda,
                                               const _Float16* __restrict__ Wt,
                                               const _Float16* __restrict__ Gt,
                                               int lane, v8f (&h)[NT], v8f (&g)[NT]) {
    const int col = lane & 15;
    const int kh  = (lane >> 4) * 8;
    const int kb  = (lane >> 4) * 8;
#pragma unroll
    for (int nt = 0; nt < NT; ++nt)
#pragma unroll
        for (int j = 0; j < 8; ++j) { h[nt][j] = 0.0f; g[nt][j] = 0.0f; }
#pragma unroll
    for (int kc = 0; kc < K / 32; ++kc) {
        Frag16 a = load_a(A + (lane & 15) * lda + kc * 32, kh);
#pragma unroll
        for (int nt = 0; nt < NT; ++nt) {
            Frag16 bw = load_b(Wt + (nt * 16 + col) * K + kc * 32 + kb);
            h[nt] = __builtin_amdgcn_wmma_f32_16x16x32_f16(
                false, a.v, false, bw.v, (short)0, h[nt], false, false);
        }
#pragma unroll
        for (int nt = 0; nt < NT; ++nt) {
            Frag16 bg = load_b(Gt + (nt * 16 + col) * K + kc * 32 + kb);
            g[nt] = __builtin_amdgcn_wmma_f32_16x16x32_f16(
                false, a.v, false, bg.v, (short)0, g[nt], false, false);
        }
    }
}

// bias + silu, store f16 activation tile [16][NT*16] to LDS.
template <int NT>
__device__ __forceinline__ void act_store(_Float16* dst, int ldd,
                                          const float* __restrict__ bias, int lane,
                                          const v8f (&acc)[NT]) {
    const int col0 = lane & 15;
    const int mb   = (lane >> 4) * 8;
#pragma unroll
    for (int nt = 0; nt < NT; ++nt) {
        float bv = bias[nt * 16 + col0];
#pragma unroll
        for (int j = 0; j < 8; ++j) {
            float y = siluf_(acc[nt][j] + bv);
            dst[(mb + j) * ldd + nt * 16 + col0] = (_Float16)y;
        }
    }
}

extern "C" __global__ __launch_bounds__(256)
void m3gnet_edge_gated_mlp(const float* __restrict__ atom,
                           const float* __restrict__ edge_attr,
                           const int*   __restrict__ eidx,
                           const float* __restrict__ prime,
                           const float* __restrict__ W0, const float* __restrict__ W1,
                           const float* __restrict__ W2,
                           const float* __restrict__ B0, const float* __restrict__ B1,
                           const float* __restrict__ B2,
                           const float* __restrict__ G0, const float* __restrict__ G1,
                           const float* __restrict__ G2,
                           const float* __restrict__ GB0, const float* __restrict__ GB1,
                           const float* __restrict__ GB2,
                           const float* __restrict__ EW, const float* __restrict__ EB,
                           float* __restrict__ out) {
    extern __shared__ char smem[];
    _Float16* w0t = (_Float16*)(smem + OFF_W0T);
    _Float16* g0t = (_Float16*)(smem + OFF_G0T);
    _Float16* w1t = (_Float16*)(smem + OFF_W1T);
    _Float16* g1t = (_Float16*)(smem + OFF_G1T);
    _Float16* w2t = (_Float16*)(smem + OFF_W2T);
    _Float16* g2t = (_Float16*)(smem + OFF_G2T);
    float* b0s  = (float*)(smem + OFF_B0);
    float* gb0s = (float*)(smem + OFF_GB0);
    float* b1s  = (float*)(smem + OFF_B1);
    float* gb1s = (float*)(smem + OFF_GB1);
    float* b2s  = (float*)(smem + OFF_B2);
    float* gb2s = (float*)(smem + OFF_GB2);
    float* ewl  = (float*)(smem + OFF_EW);
    float* ebl  = (float*)(smem + OFF_EB);

    const int tid = threadIdx.x, nth = blockDim.x;

    // One-time: transpose+convert all weights into LDS (229 KB resident).
    load_wt<320, 128>(w0t, W0, tid, nth);
    load_wt<320, 128>(g0t, G0, tid, nth);
    load_wt<128, 64 >(w1t, W1, tid, nth);
    load_wt<128, 64 >(g1t, G1, tid, nth);
    load_wt<64,  128>(w2t, W2, tid, nth);
    load_wt<64,  128>(g2t, G2, tid, nth);
    load_vec(b0s,  B0, 128, tid, nth);
    load_vec(gb0s, GB0, 128, tid, nth);
    load_vec(b1s,  B1, 64, tid, nth);
    load_vec(gb1s, GB1, 64, tid, nth);
    load_vec(b2s,  B2, 128, tid, nth);
    load_vec(gb2s, GB2, 128, tid, nth);
    load_vec(ewl,  EW, 128, tid, nth);
    if (tid == 0) ebl[0] = EB[0];
    __syncthreads();

    const int lane = tid & 31;
    const int wave = tid >> 5;
    char* stg = smem + OFF_STG + wave * STG_SIZE;
    _Float16* feat = (_Float16*)stg;            // 16x320 f16 (layer0 input)
    _Float16* h1s  = (_Float16*)stg;            // 16x128 f16 (reuses feat)
    _Float16* g1s  = (_Float16*)(stg + 4096);   // 16x128 f16
    _Float16* h2s  = (_Float16*)(stg + 8192);   // 16x64  f16
    _Float16* g2s  = (_Float16*)stg;            // 16x64  f16 (reuses h1s)
    float*    rowf = (float*)stg;               // 16x128 f32 result transpose
    float*    sarr = (float*)(stg + 10240);     // 16 per-edge scalar gates
    const float ebv  = ebl[0];
    const float4 ew4 = *(const float4*)(ewl + lane * 4);

    for (int tile = blockIdx.x * WAVES_PER_BLOCK + wave; tile < TILES;
         tile += gridDim.x * WAVES_PER_BLOCK) {
        const int e0 = tile * 16;

        // Edge indices: one lane-parallel load, shfl-broadcast per row.
        const int idxv = (lane < 16) ? eidx[e0 + lane]
                                     : eidx[N_EDGES + e0 + (lane - 16)];

        // ---- Stage features: [atom[src] | atom[dst] | prime] -> f16 LDS ----
        {
            const int i4 = lane * 4;   // covers 0..127 per pass (512B coalesced)
            const int i2 = lane * 2;   // covers 0..63  per pass
#pragma unroll 1
            for (int r = 0; r < 16; ++r) {
                const int src = __shfl(idxv, r, 32);
                const int dst = __shfl(idxv, r + 16, 32);
                const float* s0 = atom + (long)src * 128;
                const float* s1 = atom + (long)dst * 128;
                const float* s2 = prime + (long)(e0 + r) * 64;
                _Float16* fr = feat + r * 320;
                store4h(fr + i4,       *(const float4*)(s0 + i4));
                store4h(fr + 128 + i4, *(const float4*)(s1 + i4));
                store2h(fr + 256 + i2, *(const float2*)(s2 + i2));
            }
        }

        // ---- Per-edge scalar gate: s = edge_attr . ew + eb -----------------
#pragma unroll 1
        for (int r = 0; r < 16; ++r) {
            const float4 ea = *(const float4*)(edge_attr + (long)(e0 + r) * 128 + lane * 4);
            float p = ea.x * ew4.x + ea.y * ew4.y + ea.z * ew4.z + ea.w * ew4.w;
            for (int off = 16; off > 0; off >>= 1) p += __shfl_down(p, off, 32);
            if (lane == 0) sarr[r] = p + ebv;
        }

        // ---- Layer 0: 320 -> 128, shared-A dual gemm, silu both ------------
        {
            v8f h[8], g[8];
            gemm_tile_dual<320, 8>(feat, 320, w0t, g0t, lane, h, g);
            act_store<8>(h1s, 128, b0s, lane, h);
            act_store<8>(g1s, 128, gb0s, lane, g);
        }
        // ---- Layer 1: 128 -> 64, silu both ---------------------------------
        {
            v8f h[4], g[4];
            gemm_tile<128, 4>(h1s, 128, w1t, lane, h);
            gemm_tile<128, 4>(g1s, 128, g1t, lane, g);
            act_store<4>(h2s, 64, b1s, lane, h);
            act_store<4>(g2s, 64, gb1s, lane, g);
        }
        // ---- Layer 2: 64 -> 128, silu(main)*sigmoid(gate), epilogue --------
        {
            v8f h[8], g[8];
            gemm_tile<64, 8>(h2s, 64, w2t, lane, h);
            gemm_tile<64, 8>(g2s, 64, g2t, lane, g);
            const int col0 = lane & 15;
            const int mb   = (lane >> 4) * 8;
            // Scatter h*g into LDS row-major f32 tile (after all frag reads).
#pragma unroll
            for (int nt = 0; nt < 8; ++nt) {
                float bh = b2s[nt * 16 + col0];
                float bg = gb2s[nt * 16 + col0];
#pragma unroll
                for (int j = 0; j < 8; ++j) {
                    float o = siluf_(h[nt][j] + bh) * sigmoidf_(g[nt][j] + bg);
                    rowf[(mb + j) * 128 + nt * 16 + col0] = o;
                }
            }
            // Coalesced pass: one 512B row per iteration (32 lanes x float4).
#pragma unroll 1
            for (int r = 0; r < 16; ++r) {
                const long base = (long)(e0 + r) * 128 + lane * 4;
                const float4 res = *(const float4*)(edge_attr + base);
                const float4 v   = *(const float4*)(rowf + r * 128 + lane * 4);
                const float  sc  = sarr[r];
                float4 o;
                o.x = fmaf(v.x, sc, res.x);
                o.y = fmaf(v.y, sc, res.y);
                o.z = fmaf(v.z, sc, res.z);
                o.w = fmaf(v.w, sc, res.w);
                *(float4*)(out + base) = o;
            }
        }
    }
}

extern "C" void kernel_launch(void* const* d_in, const int* in_sizes, int n_in,
                              void* d_out, int out_size, void* d_ws, size_t ws_size,
                              hipStream_t stream) {
    (void)in_sizes; (void)n_in; (void)d_ws; (void)ws_size; (void)out_size;
    const float* atom  = (const float*)d_in[0];
    const float* eattr = (const float*)d_in[1];
    const int*   eidx  = (const int*)d_in[2];
    const float* prime = (const float*)d_in[3];
    const float* W0 = (const float*)d_in[4];
    const float* W1 = (const float*)d_in[5];
    const float* W2 = (const float*)d_in[6];
    const float* B0 = (const float*)d_in[7];
    const float* B1 = (const float*)d_in[8];
    const float* B2 = (const float*)d_in[9];
    const float* G0 = (const float*)d_in[10];
    const float* G1 = (const float*)d_in[11];
    const float* G2 = (const float*)d_in[12];
    const float* GB0 = (const float*)d_in[13];
    const float* GB1 = (const float*)d_in[14];
    const float* GB2 = (const float*)d_in[15];
    const float* EW = (const float*)d_in[16];
    const float* EB = (const float*)d_in[17];
    float* out = (float*)d_out;

    dim3 grid(512), block(256);   // persistent blocks; 8 waves each
    m3gnet_edge_gated_mlp<<<grid, block, SMEM_TOTAL, stream>>>(
        atom, eattr, eidx, prime,
        W0, W1, W2, B0, B1, B2,
        G0, G1, G2, GB0, GB1, GB2,
        EW, EB, out);
}
